// MultiStepSNNResidualBlock_35046933135761
// MI455X (gfx1250) — matrix operand order repaired and checked by reading, not compiled, not run on hardware
//
#include <hip/hip_runtime.h>
#include <hip/hip_bf16.h>
#include <math.h>

// ---------------------------------------------------------------------------
// MultiStepSNNResidualBlock for MI455X (gfx1250, wave32, WMMA bf16 + TDM)
// T=4, B=16, C=256, L=2048, O=2C=512
// ---------------------------------------------------------------------------

typedef __bf16 bf16_t;
typedef __attribute__((ext_vector_type(16))) __bf16 v16bf;
typedef __attribute__((ext_vector_type(8)))  float  v8f;

#define T_DIM 4
#define B_DIM 16
#define C_DIM 256
#define L_DIM 2048
#define O_DIM 512
#define LT 32            // L tile per block
#define NK_CONV 24       // 768 / 32
#define NK_LIN  8        // 256 / 32
#define YST 36           // ytile / zbuf row stride (bf16 elems)
#define SST 36           // spike buffer row stride
#define XST 34           // staged x tile row stride (f32 elems, dense TDM layout)

#if defined(__has_builtin)
#if __has_builtin(__builtin_amdgcn_tensor_load_to_lds) && \
    __has_builtin(__builtin_amdgcn_s_wait_tensorcnt)
#define USE_TDM 1
#endif
#endif
#ifndef USE_TDM
#define USE_TDM 0
#endif

// ---------------------------------------------------------------------------
// Kernel 0: tiny embedding MLP -> proj[b][c]  (B=16 x C=256)
// ---------------------------------------------------------------------------
__global__ __launch_bounds__(256) void k_proj(
    const int* __restrict__ dstep,
    const float* __restrict__ w_emb1, const float* __restrict__ b_emb1,
    const float* __restrict__ w_emb2, const float* __restrict__ b_emb2,
    const float* __restrict__ w_projW, const float* __restrict__ b_projB,
    float* __restrict__ proj)
{
    __shared__ float hbuf[B_DIM][C_DIM];
    __shared__ float ebuf[B_DIM][C_DIM];
    const int c = threadIdx.x;
    for (int b = 0; b < B_DIM; ++b) {
        float ds = (float)dstep[b];
        float u = ds * w_emb1[c] + b_emb1[c];
        hbuf[b][c] = u / (1.0f + expf(-u));          // SiLU
    }
    __syncthreads();
    for (int b = 0; b < B_DIM; ++b) {
        float s = b_emb2[c];
        for (int j = 0; j < C_DIM; ++j) s += hbuf[b][j] * w_emb2[c * C_DIM + j];
        ebuf[b][c] = s;
    }
    __syncthreads();
    for (int b = 0; b < B_DIM; ++b) {
        float s = b_projB[c];
        for (int j = 0; j < C_DIM; ++j) s += ebuf[b][j] * w_projW[c * C_DIM + j];
        proj[b * C_DIM + c] = s;
    }
}

// ---------------------------------------------------------------------------
// A-fragment K interleave for v_wmma_*_16x16x32 (16-bit A layout, ISA 7.12.2)
// ---------------------------------------------------------------------------
__device__ __forceinline__ int a_frag_k(int lane, int e)
{
    int g = lane >> 4;
    int j = e >> 1;
    int h = e & 1;
    return 2 * (j & 3) + h + 16 * (j >> 2) + 8 * g;
}

// Repack conv weights (O=512, C=256, K=3) with kk = kslice*256 + c ordering.
__global__ __launch_bounds__(256) void k_pack_conv(
    const float* __restrict__ w, bf16_t* __restrict__ dst)
{
    int tid = blockIdx.x * 256 + threadIdx.x;        // 0 .. 393215
    int e    = tid & 15;
    int lane = (tid >> 4) & 31;
    int kt   = (tid >> 9) % NK_CONV;
    int mt   = tid / (512 * NK_CONV);
    int M    = mt * 16 + (lane & 15);
    int K    = kt * 32 + a_frag_k(lane, e);
    int ks   = K >> 8;                                // kernel tap 0..2
    int c    = K & 255;
    dst[tid] = (bf16_t)w[M * 768 + c * 3 + ks];
}

// Repack a 256x256 pointwise weight (w_skip / w_res).
__global__ __launch_bounds__(256) void k_pack_lin(
    const float* __restrict__ w, bf16_t* __restrict__ dst)
{
    int tid = blockIdx.x * 256 + threadIdx.x;        // 0 .. 65535
    int e    = tid & 15;
    int lane = (tid >> 4) & 31;
    int kt   = (tid >> 9) & 7;
    int mt   = tid >> 12;
    int M    = mt * 16 + (lane & 15);
    int K    = kt * 32 + a_frag_k(lane, e);
    dst[tid] = (bf16_t)w[M * 256 + K];
}

#if USE_TDM
// ---------------------------------------------------------------------------
// TDM: async DMA of one (256 x 34) f32 tile (row stride 2048) into LDS.
// D# per CDNA5 ISA: group0 = {count/lds_addr/global_addr/type},
// group1 = {data_size, tensor_dim0/1, tile_dim0/1/2, tensor_dim0_stride}.
// tensor_dim0 = elements remaining to the row end -> right-edge OOB zero-fill.
// ---------------------------------------------------------------------------
__device__ __forceinline__ void tdm_load_xtile(const float* gptr,
                                               unsigned lds_off,
                                               unsigned tensor_dim0)
{
    typedef unsigned int u32x4 __attribute__((ext_vector_type(4)));
    typedef int i32x4 __attribute__((ext_vector_type(4)));
    typedef int i32x8 __attribute__((ext_vector_type(8)));
    unsigned long long ga = (unsigned long long)(uintptr_t)gptr;
    u32x4 g0;
    g0[0] = 1u;                                           // count=1 (valid)
    g0[1] = lds_off;                                      // LDS byte address
    g0[2] = (unsigned)ga;                                 // global_addr[31:0]
    g0[3] = (unsigned)((ga >> 32) & 0x01ffffffull)        // global_addr[56:32]
          | (2u << 30);                                   // type = 2 ("image")
    i32x8 g1;
    g1[0] = (int)(2u << 16);                              // data_size = 4B
    g1[1] = (int)((tensor_dim0 & 0xffffu) << 16);         // tensor_dim0[15:0]
    g1[2] = (int)((tensor_dim0 >> 16) | (256u << 16));    // td0[31:16] | td1
    g1[3] = (int)((unsigned)XST << 16);                   // tile_dim0 = 34
    g1[4] = 256;                                          // tile_dim1=256, tile_dim2=0
    g1[5] = 2048;                                         // tensor_dim0_stride lo
    g1[6] = 0;
    g1[7] = 0;
    i32x4 gz = {0, 0, 0, 0};
#if __clang_major__ >= 23
    i32x8 gz8 = {0, 0, 0, 0, 0, 0, 0, 0};
    __builtin_amdgcn_tensor_load_to_lds(g0, g1, gz, gz, gz8, 0);
#else
    __builtin_amdgcn_tensor_load_to_lds(g0, g1, gz, gz, 0);
#endif
}
#endif

// ---------------------------------------------------------------------------
// Main fused kernel. grid = B*(L/LT) = 1024 blocks, 256 threads (8 waves).
// ---------------------------------------------------------------------------
__global__ __launch_bounds__(256) void k_main(
    const float* __restrict__ x, const float* __restrict__ proj,
    const bf16_t* __restrict__ Aconv,
    const bf16_t* __restrict__ Askip, const bf16_t* __restrict__ Ares,
    const float* __restrict__ b_conv,
    const float* __restrict__ b_skip, const float* __restrict__ b_res,
    float* __restrict__ out_res, float* __restrict__ out_skip)
{
    // union arena: f32 x-stage [256][34] (phase A) aliases spike buf (phase C/D)
    __shared__ __align__(16) char uni[O_DIM * SST * 2];   // 36864 B
    __shared__ bf16_t arena[C_DIM * YST];                 // ytile / zbuf alias
    __shared__ float  bconvL[O_DIM];
    __shared__ float  bskipL[C_DIM];
    __shared__ float  bresL[C_DIM];
    __shared__ float  projL[C_DIM];

    float*  xstage = (float*)uni;     // [256][XST]
    bf16_t* spk    = (bf16_t*)uni;    // [512][SST]

    const int tid  = threadIdx.x;
    const int lane = tid & 31;
    const int wave = tid >> 5;
    const int hi   = lane >> 4;      // which C/D half (M offset +8)
    const int nl   = lane & 15;      // N within tile

    const int blk = blockIdx.x;
    const int b   = blk >> 6;        // 64 L-tiles per batch row
    const int l0  = (blk & 63) * LT;

    bconvL[tid]        = b_conv[tid];
    bconvL[tid + 256]  = b_conv[tid + 256];
    bskipL[tid] = b_skip[tid];
    bresL[tid]  = b_res[tid];
    projL[tid]  = proj[b * C_DIM + tid];

#if USE_TDM
    const int shift        = (l0 == 0) ? 1 : 0;   // can't start one before row 0
    const int gstart       = l0 - 1 + shift;
    const unsigned td0     = (unsigned)(L_DIM - gstart);
    const unsigned ldsXoff = (unsigned)(uintptr_t)(void*)xstage; // low32 = LDS offset
    if (tid < 32)                                 // one TDM op, issued by wave 0
        tdm_load_xtile(x + ((size_t)(0 * B_DIM + b) * C_DIM) * L_DIM + gstart,
                       ldsXoff, td0);
#endif

    // LIF membrane state: wave owns o in [wave*64, wave*64+64), n in [0,32)
    v8f vst[4][2];
#pragma unroll
    for (int mt = 0; mt < 4; ++mt)
#pragma unroll
        for (int nt = 0; nt < 2; ++nt)
#pragma unroll
            for (int i = 0; i < 8; ++i) vst[mt][nt][i] = 0.0f;

    for (int t = 0; t < T_DIM; ++t) {
#if USE_TDM
        // ---- Phase A: TDM tile arrived -> convert to bf16 ytile (+proj) ----
        if (tid < 32) __builtin_amdgcn_s_wait_tensorcnt(0);
        __syncthreads();   // TDM data visible; prev iteration's zbuf reads done
        {
            const float pv = projL[tid];
#pragma unroll 1
            for (int cc = 0; cc < LT + 2; ++cc) {
                int l = l0 + cc - 1;
                float v = (l >= 0 && l < L_DIM)
                              ? (xstage[tid * XST + cc - shift] + pv) : 0.0f;
                arena[tid * YST + cc] = (bf16_t)v;
            }
        }
#else
        __syncthreads();   // previous iteration's zbuf reads done
        // ---- Phase A: direct staged load, float4-vectorized ----
        {
            const float* xrow = x + ((size_t)((t * B_DIM + b) * C_DIM + tid)) * L_DIM;
            const float pv = projL[tid];
            arena[tid * YST + 0] =
                (bf16_t)((l0 > 0) ? (xrow[l0 - 1] + pv) : 0.0f);
#pragma unroll
            for (int kk = 0; kk < 8; ++kk) {
                float4 v4 = *(const float4*)(xrow + l0 + 4 * kk);
                arena[tid * YST + 1 + 4 * kk + 0] = (bf16_t)(v4.x + pv);
                arena[tid * YST + 1 + 4 * kk + 1] = (bf16_t)(v4.y + pv);
                arena[tid * YST + 1 + 4 * kk + 2] = (bf16_t)(v4.z + pv);
                arena[tid * YST + 1 + 4 * kk + 3] = (bf16_t)(v4.w + pv);
            }
            arena[tid * YST + 33] =
                (bf16_t)((l0 + 32 < L_DIM) ? (xrow[l0 + 32] + pv) : 0.0f);
            if (t + 1 < T_DIM)
                __builtin_prefetch(xrow + (size_t)B_DIM * C_DIM * L_DIM + l0, 0, 1);
        }
#endif
        __syncthreads();

        // ---- Phase B: conv-as-GEMM  M=512, N=32, K=768 (bf16 WMMA) ----
        v8f acc[4][2];
#pragma unroll
        for (int mt = 0; mt < 4; ++mt)
#pragma unroll
            for (int nt = 0; nt < 2; ++nt) {
                int obase = wave * 64 + mt * 16 + 8 * hi;
#pragma unroll
                for (int i = 0; i < 8; ++i) acc[mt][nt][i] = bconvL[obase + i];
            }

#pragma unroll 1
        for (int q = 0; q < NK_CONV; ++q) {
            v16bf bb[2];
#pragma unroll
            for (int nt = 0; nt < 2; ++nt)
#pragma unroll
                for (int i = 0; i < 16; ++i) {
                    int Kt = q * 32 + 8 * hi + (i < 8 ? i : i + 8);
                    int ks = Kt >> 8;                // conv tap
                    int c  = Kt & 255;
                    bb[nt][i] = arena[c * YST + ks + nt * 16 + nl];
                }

#pragma unroll
            for (int mt = 0; mt < 4; ++mt) {
                v16bf aa = *(const v16bf*)(Aconv +
                    ((size_t)(((wave * 4 + mt) * NK_CONV + q) * 32 + lane)) * 16);
#pragma unroll
                for (int nt = 0; nt < 2; ++nt)
                    acc[mt][nt] = __builtin_amdgcn_wmma_f32_16x16x32_bf16(
                        false, aa, false, bb[nt], (short)0, acc[mt][nt],
                        false, false);
            }
        }

        // ---- Phase C: LIF update (v in VGPRs), spikes -> LDS ----
#pragma unroll
        for (int mt = 0; mt < 4; ++mt)
#pragma unroll
            for (int nt = 0; nt < 2; ++nt)
#pragma unroll
                for (int i = 0; i < 8; ++i) {
                    float y2 = acc[mt][nt][i];
                    float vp = vst[mt][nt][i];
                    float v  = vp + (y2 - vp) * (1.0f / 1.2f);   // 1/TAU
                    float s  = (v - 0.5f) >= 0.0f ? 1.0f : 0.0f; // VTH
                    vst[mt][nt][i] = v * (1.0f - s);
                    int o = wave * 64 + mt * 16 + i + 8 * hi;
                    int n = nt * 16 + nl;
                    spk[o * SST + n] = (bf16_t)s;
                }
        __syncthreads();

        // ---- Phase D: z = sigmoid(gate)*tanh(filt) -> zbuf (alias arena) ----
        {
            const int c = tid;
#pragma unroll 1
            for (int n = 0; n < LT; ++n) {
                float g = (float)spk[c * SST + n];
                float f = (float)spk[(c + C_DIM) * SST + n];
                float z = (1.0f / (1.0f + expf(-g))) * tanhf(f);
                arena[c * YST + n] = (bf16_t)z;
            }
        }
        __syncthreads();   // spk dead from here -> safe to overwrite x-stage

#if USE_TDM
        // ---- Kick the next tile's DMA; it overlaps the skip/res GEMMs ----
        if (t + 1 < T_DIM && tid < 32)
            tdm_load_xtile(
                x + ((size_t)((t + 1) * B_DIM + b) * C_DIM) * L_DIM + gstart,
                ldsXoff, td0);
#endif

        // ---- Phase E: skip (waves 0-3) / res (waves 4-7) GEMMs  M=256,K=256 -
        const bf16_t* Am = (wave < 4) ? Askip : Ares;
        const float*  bL = (wave < 4) ? bskipL : bresL;
        const int     w4 = wave & 3;

        v8f acc2[4][2];
#pragma unroll
        for (int mt = 0; mt < 4; ++mt)
#pragma unroll
            for (int nt = 0; nt < 2; ++nt) {
                int obase = w4 * 64 + mt * 16 + 8 * hi;
#pragma unroll
                for (int i = 0; i < 8; ++i) acc2[mt][nt][i] = bL[obase + i];
            }

#pragma unroll 1
        for (int q = 0; q < NK_LIN; ++q) {
            v16bf bb[2];
#pragma unroll
            for (int nt = 0; nt < 2; ++nt)
#pragma unroll
                for (int i = 0; i < 16; ++i) {
                    int c = q * 32 + 8 * hi + (i < 8 ? i : i + 8);
                    bb[nt][i] = arena[c * YST + nt * 16 + nl];
                }

#pragma unroll
            for (int mt = 0; mt < 4; ++mt) {
                v16bf aa = *(const v16bf*)(Am +
                    ((size_t)(((w4 * 4 + mt) * NK_LIN + q) * 32 + lane)) * 16);
#pragma unroll
                for (int nt = 0; nt < 2; ++nt)
                    acc2[mt][nt] = __builtin_amdgcn_wmma_f32_16x16x32_bf16(
                        false, aa, false, bb[nt], (short)0, acc2[mt][nt],
                        false, false);
            }
        }

        // ---- Phase F: write skip / (x + res) ----
#pragma unroll
        for (int mt = 0; mt < 4; ++mt)
#pragma unroll
            for (int nt = 0; nt < 2; ++nt)
#pragma unroll
                for (int i = 0; i < 8; ++i) {
                    int o = w4 * 64 + mt * 16 + i + 8 * hi;
                    int l = l0 + nt * 16 + nl;
                    size_t idx = ((size_t)((t * B_DIM + b) * C_DIM + o)) * L_DIM + l;
                    if (wave < 4) out_skip[idx] = acc2[mt][nt][i];
                    else          out_res[idx]  = x[idx] + acc2[mt][nt][i];
                }
    }
}

// ---------------------------------------------------------------------------
extern "C" void kernel_launch(void* const* d_in, const int* in_sizes, int n_in,
                              void* d_out, int out_size, void* d_ws, size_t ws_size,
                              hipStream_t stream) {
    const float* x       = (const float*)d_in[0];
    const int*   dstep   = (const int*)d_in[1];
    const float* w_emb1  = (const float*)d_in[2];
    const float* b_emb1  = (const float*)d_in[3];
    const float* w_emb2  = (const float*)d_in[4];
    const float* b_emb2  = (const float*)d_in[5];
    const float* w_projW = (const float*)d_in[6];
    const float* b_projB = (const float*)d_in[7];
    const float* w_conv  = (const float*)d_in[8];
    const float* b_conv  = (const float*)d_in[9];
    const float* w_skip  = (const float*)d_in[10];
    const float* b_skip  = (const float*)d_in[11];
    const float* w_res   = (const float*)d_in[12];
    const float* b_res   = (const float*)d_in[13];

    float* out_res  = (float*)d_out;
    float* out_skip = out_res + (size_t)T_DIM * B_DIM * C_DIM * L_DIM;

    // workspace layout (256B-aligned slabs)
    char* ws = (char*)d_ws;
    float*  proj  = (float*)ws;                               // 16 KB
    bf16_t* Aconv = (bf16_t*)(ws + 16384);                    // 768 KB
    bf16_t* Askip = (bf16_t*)(ws + 16384 + 786432);           // 128 KB
    bf16_t* Ares  = (bf16_t*)(ws + 16384 + 786432 + 131072);  // 128 KB

    k_proj<<<1, 256, 0, stream>>>(dstep, w_emb1, b_emb1, w_emb2, b_emb2,
                                  w_projW, b_projB, proj);
    k_pack_conv<<<1536, 256, 0, stream>>>(w_conv, Aconv);
    k_pack_lin<<<256, 256, 0, stream>>>(w_skip, Askip);
    k_pack_lin<<<256, 256, 0, stream>>>(w_res, Ares);

    k_main<<<B_DIM * (L_DIM / LT), 256, 0, stream>>>(
        x, proj, Aconv, Askip, Ares, b_conv, b_skip, b_res, out_res, out_skip);
}
